// SelfAttentionV2Causal_21388937134112
// MI455X (gfx1250) — matrix operand order repaired and checked by reading, not compiled, not run on hardware
//
#include <hip/hip_runtime.h>

// ---------------------------------------------------------------------------
// CDNA5 (gfx1250) causal self-attention, bf16 WMMA pipeline.
// B=4, S=2048, D=1024. wave32. v_wmma_f32_16x16x32_bf16 everywhere.
// ---------------------------------------------------------------------------

typedef __bf16 bf16_t;
typedef __attribute__((ext_vector_type(16))) __bf16 v16bf;
typedef __attribute__((ext_vector_type(8)))  __bf16 bf8v;    // 16 bytes
typedef __attribute__((ext_vector_type(4)))  __bf16 bf4v;    // 8 bytes
typedef __attribute__((ext_vector_type(8)))  float  v8f;
typedef __attribute__((ext_vector_type(4)))  float  f32x4;

#define BATCH 4
#define SEQ   2048
#define DIM   1024
#define BS    (BATCH * SEQ)          // 8192

// ---- workspace layout (bytes) ---------------------------------------------
#define XB_OFF   (size_t)0                                   // x  bf16 [8192][1024]
#define WB_OFF   (XB_OFF + (size_t)BS  * DIM * 2)            // W  bf16 [3][1024][1024]
#define QB_OFF   (WB_OFF + (size_t)3   * DIM * DIM * 2)      // q  bf16 [8192][1024]
#define KB_OFF   (QB_OFF + (size_t)BS  * DIM * 2)            // k  bf16 [8192][1024]
#define VT_OFF   (KB_OFF + (size_t)BS  * DIM * 2)            // vT bf16 [4][1024][2048]
#define SC_OFF   (VT_OFF + (size_t)BS  * DIM * 2)            // S  f32  [4][2048][2048]
#define PB_OFF   (SC_OFF + (size_t)BATCH * SEQ * SEQ * 4)    // P  bf16 [4][2048][2048]

// ---------------------------------------------------------------------------
// WMMA helpers — layouts per CDNA5 ISA 7.12.2 (wave32).
// ---------------------------------------------------------------------------
__device__ __forceinline__ v8f wmma_bf16(v16bf a, v16bf b, v8f c) {
  return __builtin_amdgcn_wmma_f32_16x16x32_bf16(false, a, false, b, (short)0, c,
                                                 false, false);
}

// A-matrix 16x32 bf16, row-major source with leading dim ld.
// lane<16 : row M=lane,    K = 0..7  and 16..23
// lane>=16: row M=lane-16, K = 8..15 and 24..31
__device__ __forceinline__ v16bf load_a(const bf16_t* __restrict__ base, int ld,
                                        int row0, int k0, int lane) {
  const int m    = lane & 15;
  const int kOff = (lane >> 4) << 3;               // 0 or 8
  const bf16_t* p = base + (size_t)(row0 + m) * ld + k0 + kOff;
  union { v16bf v; bf8v h[2]; } u;
  u.h[0] = *(const bf8v*)(p);                      // K = kOff .. kOff+7
  u.h[1] = *(const bf8v*)(p + 16);                 // K = kOff+16 .. kOff+23
  return u.v;
}

// B-matrix 32x16 bf16 with B[k][n] = src[(col0+n)*ld + k0 + k]  (i.e. the
// source holds row n = one 16-contiguous-K strip -> pure contiguous loads).
// lane%16 = column N; lane<16 -> K=0..7,16..23, lane>=16 -> K=8..15,24..31.
__device__ __forceinline__ v16bf load_b(const bf16_t* __restrict__ base, int ld,
                                        int k0, int col0, int lane) {
  const int n    = lane & 15;
  const int kOff = (lane >> 4) << 3;               // 0 or 8
  const bf16_t* p = base + (size_t)(col0 + n) * ld + k0 + kOff;
  union { v16bf v; bf8v h[2]; } u;
  u.h[0] = *(const bf8v*)(p);
  u.h[1] = *(const bf8v*)(p + 16);
  return u.v;
}

// ---------------------------------------------------------------------------
// Kernel 0: f32 -> bf16 streaming convert (8 elems / thread).
// ---------------------------------------------------------------------------
__global__ __launch_bounds__(256) void cvt_kernel(const float* __restrict__ src,
                                                  bf16_t* __restrict__ dst) {
  const size_t i = ((size_t)blockIdx.x * 256 + threadIdx.x) * 8;
  f32x4 a = *(const f32x4*)(src + i);
  f32x4 b = *(const f32x4*)(src + i + 4);
  bf8v o;
  o[0] = (bf16_t)a[0]; o[1] = (bf16_t)a[1]; o[2] = (bf16_t)a[2]; o[3] = (bf16_t)a[3];
  o[4] = (bf16_t)b[0]; o[5] = (bf16_t)b[1]; o[6] = (bf16_t)b[2]; o[7] = (bf16_t)b[3];
  *(bf8v*)(dst + i) = o;
}

// ---------------------------------------------------------------------------
// Kernel 1: projections. C[m,o] = sum_d x[m,d] * W[o,d]   (bf16 -> f32 -> bf16)
// grid = (BS/128, DIM/128, 3)   block = 256 (8 waves, each 32x64 of a 128x128 tile)
// which: 0 -> q (row-major), 1 -> k (row-major), 2 -> v stored transposed vT[d][s]
// ---------------------------------------------------------------------------
__global__ __launch_bounds__(256) void proj_kernel(const bf16_t* __restrict__ xb,
                                                   const bf16_t* __restrict__ wb,
                                                   bf16_t* __restrict__ qb,
                                                   bf16_t* __restrict__ kb,
                                                   bf16_t* __restrict__ vT) {
  const int which = blockIdx.z;
  const bf16_t* W = wb + (size_t)which * DIM * DIM;
  const int wave = threadIdx.x >> 5;
  const int lane = threadIdx.x & 31;
  const int row0 = blockIdx.x * 128 + (wave >> 1) * 32;   // 2 row sub-tiles
  const int col0 = blockIdx.y * 128 + (wave & 1) * 64;    // 4 col sub-tiles

  v8f acc[2][4];
  const v8f z = {0.f, 0.f, 0.f, 0.f, 0.f, 0.f, 0.f, 0.f};
  for (int r = 0; r < 2; ++r)
    for (int c = 0; c < 4; ++c) acc[r][c] = z;

  for (int k0 = 0; k0 < DIM; k0 += 32) {
    __builtin_prefetch(xb + (size_t)(row0 + (lane & 15)) * DIM + k0 + 64, 0, 1);
    v16bf a0 = load_a(xb, DIM, row0,      k0, lane);
    v16bf a1 = load_a(xb, DIM, row0 + 16, k0, lane);
#pragma unroll
    for (int c = 0; c < 4; ++c) {
      v16bf b = load_b(W, DIM, k0, col0 + c * 16, lane);
      acc[0][c] = wmma_bf16(a0, b, acc[0][c]);
      acc[1][c] = wmma_bf16(a1, b, acc[1][c]);
    }
  }

  const int n     = lane & 15;
  const int mBase = (lane >> 4) << 3;                     // 0 or 8
  if (which < 2) {
    bf16_t* dst = (which == 0) ? qb : kb;
#pragma unroll
    for (int rt = 0; rt < 2; ++rt)
#pragma unroll
      for (int ct = 0; ct < 4; ++ct) {
        const int o = col0 + ct * 16 + n;
#pragma unroll
        for (int r = 0; r < 8; ++r)
          dst[(size_t)(row0 + rt * 16 + mBase + r) * DIM + o] =
              (bf16_t)acc[rt][ct][r];
      }
  } else {
    // vT[b][o][s]: each lane packs 8 contiguous s-values -> one b128 store
#pragma unroll
    for (int rt = 0; rt < 2; ++rt) {
      const int m0 = row0 + rt * 16 + mBase;              // global row in BS
      const int b  = m0 >> 11;
      const int s0 = m0 & (SEQ - 1);
#pragma unroll
      for (int ct = 0; ct < 4; ++ct) {
        const int o = col0 + ct * 16 + n;
        bf8v pk;
#pragma unroll
        for (int r = 0; r < 8; ++r) pk[r] = (bf16_t)acc[rt][ct][r];
        *(bf8v*)(vT + ((size_t)(b * DIM + o) << 11) + s0) = pk;
      }
    }
  }
}

// ---------------------------------------------------------------------------
// Kernel 2: causal scores. S[q,j] = (1/32) * <q_row, k_row>, masked to -1e30.
// grid = (16, 16, 4); upper-triangle tile blocks exit immediately.
// ---------------------------------------------------------------------------
__global__ __launch_bounds__(256) void scores_kernel(const bf16_t* __restrict__ qb,
                                                     const bf16_t* __restrict__ kb,
                                                     float* __restrict__ Sbuf) {
  const int ib = blockIdx.x, jb = blockIdx.y, bat = blockIdx.z;
  if (jb > ib) return;                                    // causally dead block
  const int wave = threadIdx.x >> 5;
  const int lane = threadIdx.x & 31;
  const int row0 = ib * 128 + (wave >> 1) * 32;           // query rows (in-batch)
  const int col0 = jb * 128 + (wave & 1) * 64;            // key cols  (in-batch)
  const bf16_t* Q = qb + (size_t)bat * SEQ * DIM;
  const bf16_t* K = kb + (size_t)bat * SEQ * DIM;
  float* S = Sbuf + (size_t)bat * SEQ * SEQ;

  v8f acc[2][4];
  const v8f z = {0.f, 0.f, 0.f, 0.f, 0.f, 0.f, 0.f, 0.f};
  for (int r = 0; r < 2; ++r)
    for (int c = 0; c < 4; ++c) acc[r][c] = z;

  for (int k0 = 0; k0 < DIM; k0 += 32) {
    __builtin_prefetch(Q + (size_t)(row0 + (lane & 15)) * DIM + k0 + 64, 0, 1);
    v16bf a0 = load_a(Q, DIM, row0,      k0, lane);
    v16bf a1 = load_a(Q, DIM, row0 + 16, k0, lane);
#pragma unroll
    for (int c = 0; c < 4; ++c) {
      v16bf b = load_b(K, DIM, k0, col0 + c * 16, lane);
      acc[0][c] = wmma_bf16(a0, b, acc[0][c]);
      acc[1][c] = wmma_bf16(a1, b, acc[1][c]);
    }
  }

  const float scale = 0.03125f;                           // 1/sqrt(1024)
  const int n     = lane & 15;
  const int mBase = (lane >> 4) << 3;
#pragma unroll
  for (int rt = 0; rt < 2; ++rt)
#pragma unroll
    for (int ct = 0; ct < 4; ++ct) {
      const int gk = col0 + ct * 16 + n;
#pragma unroll
      for (int r = 0; r < 8; ++r) {
        const int gq = row0 + rt * 16 + mBase + r;
        const float v = (gk <= gq) ? acc[rt][ct][r] * scale : -1.0e30f;
        S[(size_t)gq * SEQ + gk] = v;
      }
    }
}

// ---------------------------------------------------------------------------
// Kernel 3: row softmax (one wave32 per row, causally clipped length).
// Emits P in bf16; masked diagonal entries exp() to exact zero.
// ---------------------------------------------------------------------------
__global__ __launch_bounds__(256) void softmax_kernel(const float* __restrict__ Sbuf,
                                                      bf16_t* __restrict__ P) {
  const int wave = threadIdx.x >> 5;
  const int lane = threadIdx.x & 31;
  const int gid  = blockIdx.x * 8 + wave;                 // row in [0, B*S)
  const int i    = gid & (SEQ - 1);
  const int Jc   = ((i >> 7) + 1) << 7;                   // causal tile bound
  const float* Srow = Sbuf + ((size_t)gid << 11);
  bf16_t* Prow = P + ((size_t)gid << 11);

  float mx = -3.0e38f;
  for (int j = lane * 4; j < Jc; j += 128) {
    f32x4 s = *(const f32x4*)(Srow + j);
    mx = fmaxf(mx, fmaxf(fmaxf(s[0], s[1]), fmaxf(s[2], s[3])));
  }
#pragma unroll
  for (int off = 16; off > 0; off >>= 1) mx = fmaxf(mx, __shfl_xor(mx, off));

  float sum = 0.f;
  for (int j = lane * 4; j < Jc; j += 128) {
    f32x4 s = *(const f32x4*)(Srow + j);
    sum += __expf(s[0] - mx) + __expf(s[1] - mx) +
           __expf(s[2] - mx) + __expf(s[3] - mx);
  }
#pragma unroll
  for (int off = 16; off > 0; off >>= 1) sum += __shfl_xor(sum, off);
  const float inv = 1.0f / sum;

  for (int j = lane * 4; j < Jc; j += 128) {
    f32x4 s = *(const f32x4*)(Srow + j);
    bf4v o;
    o[0] = (bf16_t)(__expf(s[0] - mx) * inv);
    o[1] = (bf16_t)(__expf(s[1] - mx) * inv);
    o[2] = (bf16_t)(__expf(s[2] - mx) * inv);
    o[3] = (bf16_t)(__expf(s[3] - mx) * inv);
    *(bf4v*)(Prow + j) = o;
  }
}

// ---------------------------------------------------------------------------
// Kernel 4: O = P @ V via vT.  K-loop causally clipped to (ib+1)*128.
// grid = (16, 8, 4). Output written f32 straight to d_out.
// ---------------------------------------------------------------------------
__global__ __launch_bounds__(256) void out_kernel(const bf16_t* __restrict__ P,
                                                  const bf16_t* __restrict__ vT,
                                                  float* __restrict__ out) {
  const int ib = blockIdx.x, nb = blockIdx.y, bat = blockIdx.z;
  const int wave = threadIdx.x >> 5;
  const int lane = threadIdx.x & 31;
  const int row0 = ib * 128 + (wave >> 1) * 32;           // query rows (in-batch)
  const int col0 = nb * 128 + (wave & 1) * 64;            // output dims
  const bf16_t* Pb = P  + (size_t)bat * SEQ * SEQ;
  const bf16_t* Vb = vT + (size_t)bat * DIM * SEQ;
  float* Ob = out + (size_t)bat * SEQ * DIM;

  v8f acc[2][4];
  const v8f z = {0.f, 0.f, 0.f, 0.f, 0.f, 0.f, 0.f, 0.f};
  for (int r = 0; r < 2; ++r)
    for (int c = 0; c < 4; ++c) acc[r][c] = z;

  const int kMax = (ib + 1) * 128;                        // causal clip
  for (int j0 = 0; j0 < kMax; j0 += 32) {
    __builtin_prefetch(Pb + (size_t)(row0 + (lane & 15)) * SEQ + j0 + 64, 0, 1);
    v16bf a0 = load_a(Pb, SEQ, row0,      j0, lane);
    v16bf a1 = load_a(Pb, SEQ, row0 + 16, j0, lane);
#pragma unroll
    for (int c = 0; c < 4; ++c) {
      // B[j][n] = vT[(col0+c*16+n)][j] -> contiguous strips of vT rows
      v16bf b = load_b(Vb, SEQ, j0, col0 + c * 16, lane);
      acc[0][c] = wmma_bf16(a0, b, acc[0][c]);
      acc[1][c] = wmma_bf16(a1, b, acc[1][c]);
    }
  }

  const int n     = lane & 15;
  const int mBase = (lane >> 4) << 3;
#pragma unroll
  for (int rt = 0; rt < 2; ++rt)
#pragma unroll
    for (int ct = 0; ct < 4; ++ct) {
      const int d = col0 + ct * 16 + n;
#pragma unroll
      for (int r = 0; r < 8; ++r)
        Ob[(size_t)(row0 + rt * 16 + mBase + r) * DIM + d] = acc[rt][ct][r];
    }
}

// ---------------------------------------------------------------------------
extern "C" void kernel_launch(void* const* d_in, const int* in_sizes, int n_in,
                              void* d_out, int out_size, void* d_ws, size_t ws_size,
                              hipStream_t stream) {
  (void)in_sizes; (void)n_in; (void)out_size; (void)ws_size;
  const float* x  = (const float*)d_in[0];
  const float* Wq = (const float*)d_in[1];
  const float* Wk = (const float*)d_in[2];
  const float* Wv = (const float*)d_in[3];
  float* out = (float*)d_out;

  char* ws = (char*)d_ws;
  bf16_t* xb = (bf16_t*)(ws + XB_OFF);
  bf16_t* wb = (bf16_t*)(ws + WB_OFF);
  bf16_t* qb = (bf16_t*)(ws + QB_OFF);
  bf16_t* kb = (bf16_t*)(ws + KB_OFF);
  bf16_t* vT = (bf16_t*)(ws + VT_OFF);
  float*  Sb = (float*) (ws + SC_OFF);
  bf16_t* Pb = (bf16_t*)(ws + PB_OFF);

  // 0) f32 -> bf16 conversions
  cvt_kernel<<<(BS * DIM) / 2048, 256, 0, stream>>>(x, xb);
  cvt_kernel<<<(DIM * DIM) / 2048, 256, 0, stream>>>(Wq, wb);
  cvt_kernel<<<(DIM * DIM) / 2048, 256, 0, stream>>>(Wk, wb + (size_t)DIM * DIM);
  cvt_kernel<<<(DIM * DIM) / 2048, 256, 0, stream>>>(Wv, wb + (size_t)2 * DIM * DIM);

  // 1) q, k, vT projections
  proj_kernel<<<dim3(BS / 128, DIM / 128, 3), 256, 0, stream>>>(xb, wb, qb, kb, vT);

  // 2) causal scores (lower-triangle tile blocks only)
  scores_kernel<<<dim3(SEQ / 128, SEQ / 128, BATCH), 256, 0, stream>>>(qb, kb, Sb);

  // 3) row softmax -> P (bf16)
  softmax_kernel<<<BS / 8, 256, 0, stream>>>(Sb, Pb);

  // 4) O = P @ V
  out_kernel<<<dim3(SEQ / 128, DIM / 128, BATCH), 256, 0, stream>>>(Pb, vT, out);
}
